// HMM_33217277067556
// MI455X (gfx1250) — matrix operand order repaired and checked by reading, not compiled, not run on hardware
//
#include <hip/hip_runtime.h>

#define S_DIM 2048
#define V_DIM 32768
#define T_DIM 4096

typedef __bf16 bf16_t;
typedef __attribute__((ext_vector_type(16))) __bf16 v16bf;
typedef __attribute__((ext_vector_type(8)))  float  v8f;
typedef unsigned int u32x4 __attribute__((ext_vector_type(4)));
typedef int          i32x4 __attribute__((ext_vector_type(4)));
typedef int          i32x8 __attribute__((ext_vector_type(8)));

union V8FU  { v8f  v; float  f[8];  };
union V16BU { v16bf v; bf16_t h[16]; };

#if defined(__has_builtin)
#if __has_builtin(__builtin_amdgcn_tensor_load_to_lds)
#define HMM_HAVE_TDM 1
#endif
#endif

// ---------------------------------------------------------------------------
// Prep: AT[j][k] = (bf16) A[k][j]   (8 MB, stays L2-resident for the scan)
// ---------------------------------------------------------------------------
__global__ void hmm_prep_at(const float* __restrict__ A, bf16_t* __restrict__ AT) {
    const int idx = blockIdx.x * blockDim.x + threadIdx.x;
    const int k = idx & (S_DIM - 1);
    const int j = idx >> 11;
    AT[(size_t)j * S_DIM + k] = (bf16_t)A[(size_t)k * S_DIM + j];
}

// ---------------------------------------------------------------------------
// Prep: B_obs[t][s] = B[s][obs[t]]  (33 MB; one pass instead of 4096 column
// gathers from the 256 MB emission matrix during the scan)
// ---------------------------------------------------------------------------
__global__ void hmm_gather_b(const float* __restrict__ B, const int* __restrict__ obs,
                             float* __restrict__ Bobs) {
    const int idx = blockIdx.x * blockDim.x + threadIdx.x;
    const int s = idx & (S_DIM - 1);
    const int t = idx >> 11;           // one t per 8 blocks of 256 threads (2048/256)
    const int o = obs[t];
    Bobs[idx] = B[(size_t)s * V_DIM + o];
}

// ---------------------------------------------------------------------------
// Init: alpha0 = pi * B_obs[0]; reset barrier counter (graph-replay safe)
// ---------------------------------------------------------------------------
__global__ void hmm_init(const float* __restrict__ pi, const float* __restrict__ Bobs,
                         float* __restrict__ aF, bf16_t* __restrict__ aB,
                         unsigned* __restrict__ cnt) {
    const int s = blockIdx.x * blockDim.x + threadIdx.x;
    const float v = pi[s] * Bobs[s];
    aF[s] = v;
    aB[s] = (bf16_t)v;
    if (s == 0) *cnt = 0u;
}

// ---------------------------------------------------------------------------
// One WMMA partial step: acc += x_rep(16x32) * AT-tile(32x16), fp32 accum.
// A-frag (16x32 bf16, contiguous K, same as B): lanes 0-15 K=0..15, lanes 16-31 K=16..31;
// all 16 M-rows replicated with the alpha vector (matvec via WMMA).
// B-frag (32x16 bf16): lane = column n, lanes 16-31 hold K=16..31 -> one
// contiguous 32-byte load from transposed AT row.
// ---------------------------------------------------------------------------
__device__ __forceinline__ v8f wstep(const bf16_t* __restrict__ xl,
                                     const bf16_t* __restrict__ bptr,
                                     int k0, int s8, v8f c) {
    V16BU a;
#pragma unroll
    for (int e = 0; e < 8; ++e) {
        a.h[e]     = xl[k0 + 2 * s8 + e];      // K = 2*s8 + e
        a.h[8 + e] = xl[k0 + 2 * s8 + 8 + e];  // K = 2*s8 + 8 + e
    }
    const v16bf b = *reinterpret_cast<const v16bf*>(bptr);  // 32B contiguous
    return __builtin_amdgcn_wmma_f32_16x16x32_bf16(
        /*neg_a=*/false, a.v, /*neg_b=*/false, b,
        /*c_mod=*/(short)0, c, /*reuse_a=*/false, /*reuse_b=*/false);
}

// ---------------------------------------------------------------------------
// Persistent scan kernel: 32 blocks x 128 threads (4 waves); each wave owns a
// 16-wide output tile. Per step: 8 MB of bf16 A streamed from L2 via WMMA,
// alpha staged into LDS by the Tensor Data Mover (wave 0 issues a 1D D#:
// 512 x 8B elements = 4 KB, completed via s_wait_tensorcnt), then a
// device-wide monotonic atomic barrier (one launch instead of 4095).
// ---------------------------------------------------------------------------
__global__ void __launch_bounds__(128) hmm_forward(const bf16_t* __restrict__ AT,
                                                   const float* __restrict__ Bobs,
                                                   float* __restrict__ aF,
                                                   bf16_t* __restrict__ aB,
                                                   unsigned* __restrict__ cnt) {
    const int tid   = threadIdx.x;
    const int lane  = tid & 31;
    const int wave  = tid >> 5;
    const int j0    = (blockIdx.x * 4 + wave) * 16;
    const int n     = lane & 15;
    const int s8    = (lane & 16) ? 8 : 0;
    const int off16 = (lane & 16) ? 16 : 0;
    const bf16_t* atrow = AT + (size_t)(j0 + n) * S_DIM + off16;
    const unsigned nb = gridDim.x;

    __shared__ bf16_t xl[S_DIM];   // 4 KB staged alpha (bf16)

    for (int t = 1; t < T_DIM; ++t) {
        const bf16_t* xin = aB + ((t - 1) & 1) * S_DIM;

#ifdef HMM_HAVE_TDM
        // Stage previous alpha into LDS via the Tensor Data Mover.
        if (tid < 32) {                          // one DMA per block (wave 0)
            const unsigned long long ga = (unsigned long long)(const void*)xin;
            const unsigned lds_off = (unsigned)(unsigned long long)(const void*)xl;
            u32x4 g0;
            g0.x = 1u;                           // count=1 (valid user D#)
            g0.y = lds_off;                      // lds_addr (bytes)
            g0.z = (unsigned)ga;                 // global_addr[31:0]
            g0.w = (unsigned)((ga >> 32) & 0x01FFFFFFull) | (2u << 30); // [56:32]|type=2
            i32x8 g1;
            g1[0] = (int)(3u << 16);                   // data_size=8B, no flags
            g1[1] = (int)((512u & 0xFFFFu) << 16);     // tensor_dim0.lo16 (=512)
            g1[2] = (int)((512u >> 16) | (1u << 16));  // tensor_dim0.hi16 | tensor_dim1=1.lo16
            g1[3] = (int)(512u << 16);                 // tensor_dim1.hi16 | tile_dim0=512
            g1[4] = 0;                                 // tile_dim1=0, tile_dim2=0
            g1[5] = (int)512u;                         // tensor_dim0_stride.lo32
            g1[6] = 0;                                 // stride0.hi16 | stride1.lo16
            g1[7] = 0;                                 // stride1.hi32
            i32x4 gz = {0, 0, 0, 0};
#if defined(__clang_major__) && __clang_major__ >= 23
            i32x8 gz8 = {0, 0, 0, 0, 0, 0, 0, 0};
            __builtin_amdgcn_tensor_load_to_lds(g0, g1, gz, gz, gz8, 0);
#else
            __builtin_amdgcn_tensor_load_to_lds(g0, g1, gz, gz, 0);
#endif
            __builtin_amdgcn_s_wait_tensorcnt(0);
        }
        __syncthreads();
#else
        for (int i = tid; i < S_DIM; i += 128) xl[i] = xin[i];
        __syncthreads();
#endif

        v8f c0 = {};
        v8f c1 = {};
#pragma unroll 4
        for (int k0 = 0; k0 < S_DIM; k0 += 64) {
            __builtin_prefetch((const void*)(atrow + k0 + 64), 0, 3);
            c0 = wstep(xl, atrow + k0,      k0,      s8, c0);
            c1 = wstep(xl, atrow + k0 + 32, k0 + 32, s8, c1);
        }

        // All 16 D-rows are identical (replicated A); lane reads its column,
        // scales by the emission factor, writes fp32 + bf16 alpha.
        if (lane < 16) {
            V8FU u0, u1; u0.v = c0; u1.v = c1;
            const float y = (u0.f[0] + u1.f[0]) * Bobs[(size_t)t * S_DIM + j0 + lane];
            aF[(t & 1) * S_DIM + j0 + lane] = y;
            aB[(t & 1) * S_DIM + j0 + lane] = (bf16_t)y;
        }

        // Device-wide barrier: monotonic counter, target = t * numBlocks.
        __threadfence();
        __syncthreads();
        if (tid == 0) {
            atomicAdd(cnt, 1u);
            const unsigned target = (unsigned)t * nb;
            while (*((volatile unsigned*)cnt) < target) {
                __builtin_amdgcn_s_sleep(2);
            }
            __threadfence();
        }
        __syncthreads();
    }
}

// ---------------------------------------------------------------------------
// Final reduction: sum(alpha_{T-1}) -> out[0]
// ---------------------------------------------------------------------------
__global__ void hmm_finalize(const float* __restrict__ alpha, float* __restrict__ out) {
    __shared__ float red[256];
    float s = 0.f;
    for (int i = threadIdx.x; i < S_DIM; i += 256) s += alpha[i];
    red[threadIdx.x] = s;
    __syncthreads();
    for (int o = 128; o > 0; o >>= 1) {
        if (threadIdx.x < o) red[threadIdx.x] += red[threadIdx.x + o];
        __syncthreads();
    }
    if (threadIdx.x == 0) out[0] = red[0];
}

// ---------------------------------------------------------------------------
// Host entry. Inputs: observations(int32,T), A(f32,SxS), B(f32,SxV), pi(f32,S)
// Workspace: AT bf16 (8MB) | B_obs f32 (32MB) | alpha f32 x2 | alpha bf16 x2 |
//            barrier counter.  (~42 MB total)
// ---------------------------------------------------------------------------
extern "C" void kernel_launch(void* const* d_in, const int* in_sizes, int n_in,
                              void* d_out, int out_size, void* d_ws, size_t ws_size,
                              hipStream_t stream) {
    (void)in_sizes; (void)n_in; (void)out_size; (void)ws_size;

    const int*   obs = (const int*)d_in[0];
    const float* A   = (const float*)d_in[1];
    const float* B   = (const float*)d_in[2];
    const float* pi  = (const float*)d_in[3];
    float*       out = (float*)d_out;

    char* w = (char*)d_ws;
    bf16_t*   AT   = (bf16_t*)w;   w += (size_t)S_DIM * S_DIM * sizeof(bf16_t);
    float*    Bobs = (float*)w;    w += (size_t)T_DIM * S_DIM * sizeof(float);
    float*    aF   = (float*)w;    w += (size_t)2 * S_DIM * sizeof(float);
    bf16_t*   aB   = (bf16_t*)w;   w += (size_t)2 * S_DIM * sizeof(bf16_t);
    unsigned* cnt  = (unsigned*)w;

    hmm_prep_at <<<(S_DIM * S_DIM) / 256, 256, 0, stream>>>(A, AT);
    hmm_gather_b<<<(T_DIM * S_DIM) / 256, 256, 0, stream>>>(B, obs, Bobs);
    hmm_init    <<<S_DIM / 256, 256, 0, stream>>>(pi, Bobs, aF, aB, cnt);
    hmm_forward <<<32, 128, 0, stream>>>(AT, Bobs, aF, aB, cnt);
    hmm_finalize<<<1, 256, 0, stream>>>(aF + ((T_DIM - 1) & 1) * S_DIM, out);
}
